// CenterAttention_6210522710385
// MI455X (gfx1250) — matrix operand order repaired and checked by reading, not compiled, not run on hardware
//
#include <hip/hip_runtime.h>

// ---------------------------------------------------------------------------
// CenterAttention for MI455X (gfx1250, wave32, WMMA)
//
// result[b,n,e] = sum_d spatial[b,d] * spec_weights[b,d,e]  (n-independent,
// since spatial_out is broadcast along n) -> compute (64,1024), broadcast.
// v_s (last third of Wspec output) is dead code in the reference.
//
// Big GEMMs: v_wmma_f32_16x16x32_bf16 with 2x4 register blocking
// (8 WMMAs per 6 fragment loads).  Softmax paths stay fp32.
// ---------------------------------------------------------------------------

typedef __attribute__((ext_vector_type(16))) __bf16 v16bf;
typedef __attribute__((ext_vector_type(8)))  float  v8f;

#define SCALE   0.125f   // 64^-0.5

// ---- workspace layout (bytes) ----
#define OFF_XBF   (0u)                         // x as bf16         (b,n,d)   32 MB
#define OFF_XTBF  (33554432u)                  // x^T as bf16       (b,d,n)   32 MB
#define OFF_WKVT  (67108864u)                  // Wkv^T bf16        (c,k)      2 MB
#define OFF_WSPT  (69206016u)                  // Wspec[:, :512]^T  (n,m)    256 KB
#define OFF_KV    (69468160u)                  // kv bf16           (b,n,1024) 32 MB
#define OFF_QSKS  (103022592u)                 // [q_s|k_s] bf16    (b,1024,512) 64 MB
#define OFF_SPAT  (170131456u)                 // spatial fp32      (b,1024)  256 KB

// ---------------------------------------------------------------------------
// WMMA fragment helpers (16x16x32 bf16, wave32; A/B share a contiguous k order, C/D per CDNA5 ISA 7.12.2)
// ---------------------------------------------------------------------------
union FragBF { v16bf v; uint4 q[2]; };

// A fragment: 16x32 tile, row-major, K contiguous along a row.
// lane<16: row=lane,    elems 0..15 -> K 0..15
// lane>=16: row=lane-16, elems 0..15 -> K 16..31
__device__ __forceinline__ v16bf load_frag_A(const __bf16* base, int rowStride) {
    int lane = threadIdx.x & 31;
    int row  = lane & 15;
    int kb   = (lane >> 4) << 4;              // 0 or 16
    const __bf16* p = base + row * rowStride + kb;
    FragBF f;
    f.q[0] = *reinterpret_cast<const uint4*>(p);
    f.q[1] = *reinterpret_cast<const uint4*>(p + 8);
    return f.v;
}

// B fragment: 32x16 tile stored column-major (one output column's K values
// contiguous).  lane<16: col=lane, K 0..15; lane>=16: col=lane-16, K 16..31.
__device__ __forceinline__ v16bf load_frag_B(const __bf16* base, int colStride) {
    int lane = threadIdx.x & 31;
    int col  = lane & 15;
    int kb   = (lane >> 4) << 4;              // 0 or 16
    const __bf16* p = base + col * colStride + kb;
    FragBF f;
    f.q[0] = *reinterpret_cast<const uint4*>(p);
    f.q[1] = *reinterpret_cast<const uint4*>(p + 8);
    return f.v;
}

// C/D: VGPR r, lanes 0-15 -> C[r][lane]; lanes 16-31 -> C[8+r][lane-16]
__device__ __forceinline__ void store_tile_bf16(__bf16* C, int ldc, v8f acc) {
    int lane = threadIdx.x & 31;
    int col  = lane & 15;
    int rb   = (lane >> 4) << 3;
#pragma unroll
    for (int r = 0; r < 8; ++r)
        C[(rb + r) * ldc + col] = (__bf16)acc[r];
}

#define WMMA_BF16(a, b, c) \
    __builtin_amdgcn_wmma_f32_16x16x32_bf16(false, (a), false, (b), (short)0, (c), false, false)

#define V8F_ZERO {0.f, 0.f, 0.f, 0.f, 0.f, 0.f, 0.f, 0.f}

// ---------------------------------------------------------------------------
// K0: conversions / transposes to bf16
// ---------------------------------------------------------------------------
__global__ __launch_bounds__(256) void k_convert_x(const float* __restrict__ x,
                                                   __bf16* __restrict__ xbf,
                                                   __bf16* __restrict__ xtbf) {
    int idx = blockIdx.x * 256 + threadIdx.x;          // 64*256*1024 total
    float v = x[idx];
    xbf[idx] = (__bf16)v;
    int d = idx & 1023;
    int n = (idx >> 10) & 255;
    int b = idx >> 18;
    xtbf[((b << 10) + d) * 256 + n] = (__bf16)v;
}

__global__ __launch_bounds__(256) void k_convert_wkv(const float* __restrict__ Wkv,
                                                     __bf16* __restrict__ wkvt) {
    int idx = blockIdx.x * 256 + threadIdx.x;          // 1024*1024
    int k = idx >> 10, c = idx & 1023;
    wkvt[c * 1024 + k] = (__bf16)Wkv[idx];
}

__global__ __launch_bounds__(256) void k_convert_wspec(const float* __restrict__ Wspec,
                                                       __bf16* __restrict__ wspt) {
    int idx = blockIdx.x * 256 + threadIdx.x;          // 512*256
    int n = idx >> 8, m = idx & 255;
    wspt[idx] = (__bf16)Wspec[m * 768 + n];            // only q_s|k_s columns used
}

// ---------------------------------------------------------------------------
// K1: kv = x @ Wkv   per batch: 256x1024 @ 1024x1024 -> bf16
// 2x4 register blocking: each wave computes a 32x64 output block.
// ---------------------------------------------------------------------------
__global__ __launch_bounds__(256) void k_kv_gemm(const __bf16* __restrict__ xbf,
                                                 const __bf16* __restrict__ wkvt,
                                                 __bf16* __restrict__ kv) {
    int wave = threadIdx.x >> 5;
    int t    = blockIdx.x * 8 + wave;                  // 64 * 8 * 16 = 8192 wave-blocks
    int b    = t >> 7;
    int rem  = t & 127;
    int mt   = rem >> 4;                               // 0..7   (32-row group)
    int nt   = rem & 15;                               // 0..15  (64-col group)
    const __bf16* A0 = xbf  + (size_t)(b * 256 + mt * 32) * 1024;
    const __bf16* B0 = wkvt + (size_t)(nt * 64) * 1024;
    v8f acc[2][4] = {{V8F_ZERO, V8F_ZERO, V8F_ZERO, V8F_ZERO},
                     {V8F_ZERO, V8F_ZERO, V8F_ZERO, V8F_ZERO}};
    for (int kk = 0; kk < 1024; kk += 32) {
        __builtin_prefetch(A0 + kk + 128, 0, 1);       // global_prefetch_b8
        __builtin_prefetch(B0 + kk + 128, 0, 1);
        v16bf a0 = load_frag_A(A0 + kk,             1024);
        v16bf a1 = load_frag_A(A0 + 16 * 1024 + kk, 1024);
#pragma unroll
        for (int j = 0; j < 4; ++j) {
            v16bf w = load_frag_B(B0 + (size_t)(j * 16) * 1024 + kk, 1024);
            acc[0][j] = WMMA_BF16(a0, w, acc[0][j]);
            acc[1][j] = WMMA_BF16(a1, w, acc[1][j]);
        }
    }
#pragma unroll
    for (int i = 0; i < 2; ++i)
#pragma unroll
        for (int j = 0; j < 4; ++j)
            store_tile_bf16(kv + (size_t)(b * 256 + mt * 32 + i * 16) * 1024
                               + nt * 64 + j * 16, 1024, acc[i][j]);
}

// ---------------------------------------------------------------------------
// K2: [q_s|k_s] = x^T @ Wspec[:, :512]  per batch: 1024x256 @ 256x512 -> bf16
// 2x4 register blocking, 32x64 per wave.
// ---------------------------------------------------------------------------
__global__ __launch_bounds__(256) void k_qsks_gemm(const __bf16* __restrict__ xtbf,
                                                   const __bf16* __restrict__ wspt,
                                                   __bf16* __restrict__ qsks) {
    int wave = threadIdx.x >> 5;
    int t    = blockIdx.x * 8 + wave;                  // 64 * 32 * 8 = 16384 wave-blocks
    int b    = t >> 8;
    int rem  = t & 255;
    int mt   = rem >> 3;                               // 0..31  (32-row group)
    int nt   = rem & 7;                                // 0..7   (64-col group)
    const __bf16* A0 = xtbf + (size_t)(b * 1024 + mt * 32) * 256;
    const __bf16* B0 = wspt + (size_t)(nt * 64) * 256;
    v8f acc[2][4] = {{V8F_ZERO, V8F_ZERO, V8F_ZERO, V8F_ZERO},
                     {V8F_ZERO, V8F_ZERO, V8F_ZERO, V8F_ZERO}};
    for (int kk = 0; kk < 256; kk += 32) {
        __builtin_prefetch(B0 + kk + 64, 0, 1);
        v16bf a0 = load_frag_A(A0 + kk,            256);
        v16bf a1 = load_frag_A(A0 + 16 * 256 + kk, 256);
#pragma unroll
        for (int j = 0; j < 4; ++j) {
            v16bf w = load_frag_B(B0 + (size_t)(j * 16) * 256 + kk, 256);
            acc[0][j] = WMMA_BF16(a0, w, acc[0][j]);
            acc[1][j] = WMMA_BF16(a1, w, acc[1][j]);
        }
    }
#pragma unroll
    for (int i = 0; i < 2; ++i)
#pragma unroll
        for (int j = 0; j < 4; ++j)
            store_tile_bf16(qsks + (size_t)(b * 1024 + mt * 32 + i * 16) * 512
                                 + nt * 64 + j * 16, 512, acc[i][j]);
}

// ---------------------------------------------------------------------------
// K3: center-token attention + spatial projection (fp32, tiny FLOPs)
// one block per batch, 512 threads
// ---------------------------------------------------------------------------
__global__ __launch_bounds__(512) void k_center_attn(const float* __restrict__ x,
                                                     const float* __restrict__ Wq,
                                                     const float* __restrict__ Wout,
                                                     const float* __restrict__ bout,
                                                     const __bf16* __restrict__ kv,
                                                     float* __restrict__ spatial) {
    __shared__ float q[512];
    __shared__ float outv[512];
    __shared__ float lg[256];
    __shared__ float red[256];
    int b = blockIdx.x, tid = threadIdx.x;

    // q = x_center @ Wq
    const float* xc = x + (size_t)(b * 256 + 128) * 1024;
    float acc = 0.f;
    for (int k = 0; k < 1024; ++k) acc += xc[k] * Wq[k * 512 + tid];
    q[tid] = acc;
    __syncthreads();

    for (int h = 0; h < 8; ++h) {
        if (tid < 256) {
            const __bf16* krow = kv + (size_t)(b * 256 + tid) * 1024 + h * 64;
            float l = 0.f;
            for (int d = 0; d < 64; ++d) l += q[h * 64 + d] * (float)krow[d];
            lg[tid] = l * SCALE;
            red[tid] = lg[tid];
        }
        __syncthreads();
        for (int s = 128; s > 0; s >>= 1) {
            if (tid < s) red[tid] = fmaxf(red[tid], red[tid + s]);
            __syncthreads();
        }
        float mx = red[0];
        __syncthreads();
        if (tid < 256) {
            float e = __expf(lg[tid] - mx);
            lg[tid] = e;
            red[tid] = e;
        }
        __syncthreads();
        for (int s = 128; s > 0; s >>= 1) {
            if (tid < s) red[tid] += red[tid + s];
            __syncthreads();
        }
        float denom = red[0];
        __syncthreads();
        if (tid < 64) {
            const __bf16* vcol = kv + (size_t)b * 256 * 1024 + 512 + h * 64 + tid;
            float o = 0.f;
            for (int n = 0; n < 256; ++n) o += lg[n] * (float)vcol[n * 1024];
            outv[h * 64 + tid] = o / denom;
        }
        __syncthreads();
    }

    // spatial = out @ Wout + bout
    for (int e = tid; e < 1024; e += 512) {
        float sacc = bout[e];
        for (int c = 0; c < 512; ++c) sacc += outv[c] * Wout[c * 1024 + e];
        spatial[b * 1024 + e] = sacc;
    }
}

// ---------------------------------------------------------------------------
// K4: spec attention.  L = SCALE * q_s @ k_s^T (WMMA), row softmax,
// final[e] = sum_d spatial[d] * P[d,e]; broadcast final over all n.
// One block (8 waves) per batch; each wave owns i-tiles it = wave, wave+8, ...
// A-fragments (q_s rows of an i-tile, full K=256) are hoisted into registers
// and reused across all 64 j-tiles: 8 WMMAs per 8 fragment loads.
// ---------------------------------------------------------------------------
__device__ __forceinline__ v8f spec_tile_hoisted(const v16bf afr[8],
                                                 const __bf16* qb, int j0) {
    const __bf16* B = qb + (size_t)j0 * 512 + 256;     // k_s rows as B columns
    v8f acc = V8F_ZERO;
#pragma unroll
    for (int k8 = 0; k8 < 8; ++k8) {
        v16bf w = load_frag_B(B + k8 * 32, 512);
        acc = WMMA_BF16(afr[k8], w, acc);
    }
#pragma unroll
    for (int r = 0; r < 8; ++r) acc[r] *= SCALE;
    return acc;
}

__global__ __launch_bounds__(256) void k_spec_attn(const __bf16* __restrict__ qsks,
                                                   const float* __restrict__ spatial,
                                                   float* __restrict__ out) {
    __shared__ float rmax[1024], rsum[1024], spat[1024], fin[1024];
    int b = blockIdx.x, tid = threadIdx.x;
    int wave = tid >> 5, lane = tid & 31;
    for (int e = tid; e < 1024; e += 256) {
        spat[e] = spatial[b * 1024 + e];
        fin[e]  = 0.f;
    }
    __syncthreads();
    const __bf16* qb = qsks + (size_t)b * 1024 * 512;

    // ---- pass 1: online row max / sum (flash style) ----
    for (int it = wave; it < 64; it += 8) {
        int i0 = it * 16;
        v16bf afr[8];
#pragma unroll
        for (int k8 = 0; k8 < 8; ++k8)
            afr[k8] = load_frag_A(qb + (size_t)i0 * 512 + k8 * 32, 512);
        float m[8], s[8];
#pragma unroll
        for (int r = 0; r < 8; ++r) { m[r] = -3.0e38f; s[r] = 0.f; }
        for (int jt = 0; jt < 64; ++jt) {
            v8f L = spec_tile_hoisted(afr, qb, jt * 16);
#pragma unroll
            for (int r = 0; r < 8; ++r) {
                float xv = L[r];
                float tm = xv;
                tm = fmaxf(tm, __shfl_xor(tm, 1, 16));
                tm = fmaxf(tm, __shfl_xor(tm, 2, 16));
                tm = fmaxf(tm, __shfl_xor(tm, 4, 16));
                tm = fmaxf(tm, __shfl_xor(tm, 8, 16));
                float nm = fmaxf(m[r], tm);
                float e  = __expf(xv - nm);
                float ts = e;
                ts += __shfl_xor(ts, 1, 16);
                ts += __shfl_xor(ts, 2, 16);
                ts += __shfl_xor(ts, 4, 16);
                ts += __shfl_xor(ts, 8, 16);
                s[r] = s[r] * __expf(m[r] - nm) + ts;
                m[r] = nm;
            }
        }
        if ((lane & 15) == 0) {
            int rb = i0 + ((lane >> 4) << 3);
#pragma unroll
            for (int r = 0; r < 8; ++r) { rmax[rb + r] = m[r]; rsum[rb + r] = s[r]; }
        }
    }
    __syncthreads();

    // ---- pass 2: recompute tiles, accumulate sum_d spatial[d]*P[d,e] ----
    for (int it = wave; it < 64; it += 8) {
        int i0 = it * 16;
        v16bf afr[8];
#pragma unroll
        for (int k8 = 0; k8 < 8; ++k8)
            afr[k8] = load_frag_A(qb + (size_t)i0 * 512 + k8 * 32, 512);
        int rb = (lane >> 4) << 3;
        float w[8], mm[8];
#pragma unroll
        for (int r = 0; r < 8; ++r) {
            int row = i0 + rb + r;
            w[r]  = spat[row] / rsum[row];
            mm[r] = rmax[row];
        }
        for (int jt = 0; jt < 64; ++jt) {
            v8f L = spec_tile_hoisted(afr, qb, jt * 16);
            float a = 0.f;
#pragma unroll
            for (int r = 0; r < 8; ++r) a += w[r] * __expf(L[r] - mm[r]);
            atomicAdd(&fin[jt * 16 + (lane & 15)], a);   // ds_add_f32
        }
    }
    __syncthreads();

    // ---- broadcast final vector to all 256 patch positions ----
    for (int n = 0; n < 256; ++n)
        for (int e = tid; e < 1024; e += 256)
            out[(size_t)(b * 256 + n) * 1024 + e] = fin[e];
}

// ---------------------------------------------------------------------------
extern "C" void kernel_launch(void* const* d_in, const int* in_sizes, int n_in,
                              void* d_out, int out_size, void* d_ws, size_t ws_size,
                              hipStream_t stream) {
    const float* x     = (const float*)d_in[0];
    const float* Wq    = (const float*)d_in[1];
    const float* Wkv   = (const float*)d_in[2];
    const float* Wout  = (const float*)d_in[3];
    const float* bout  = (const float*)d_in[4];
    const float* Wspec = (const float*)d_in[5];
    float* out = (float*)d_out;

    char* ws = (char*)d_ws;
    __bf16* xbf   = (__bf16*)(ws + OFF_XBF);
    __bf16* xtbf  = (__bf16*)(ws + OFF_XTBF);
    __bf16* wkvt  = (__bf16*)(ws + OFF_WKVT);
    __bf16* wspt  = (__bf16*)(ws + OFF_WSPT);
    __bf16* kv    = (__bf16*)(ws + OFF_KV);
    __bf16* qsks  = (__bf16*)(ws + OFF_QSKS);
    float*  spat  = (float*)(ws + OFF_SPAT);

    k_convert_x    <<<65536, 256, 0, stream>>>(x, xbf, xtbf);
    k_convert_wkv  <<<4096,  256, 0, stream>>>(Wkv, wkvt);
    k_convert_wspec<<<512,   256, 0, stream>>>(Wspec, wspt);
    k_kv_gemm      <<<1024,  256, 0, stream>>>(xbf, wkvt, kv);
    k_qsks_gemm    <<<2048,  256, 0, stream>>>(xtbf, wspt, qsks);
    k_center_attn  <<<64,    512, 0, stream>>>(x, Wq, Wout, bout, kv, spat);
    k_spec_attn    <<<64,    256, 0, stream>>>(qsks, spat, out);
}
